// Copy_32066225831966
// MI455X (gfx1250) — matrix operand order repaired and checked by reading, not compiled, not run on hardware
//
#include <hip/hip_runtime.h>

// Dilated 5-layer tanh RNN for MI455X (gfx1250, wave32, WMMA).
//   1) fuse_table:  table[v][i] = embed[v]@Wih0[i] + bih0[i]+bhh0[i]   (VOCAB=EMB=10 -> LUT)
//   2) gather_pre0: pre[r][i] = table[tok[r]][i]                        (layer-0 input GEMM = gather)
//   3) rnn_scan:    z = Whh x h^T  (A = Whh static/hoisted, B = h row-major in LDS).
//                   C element (M=i, N=b) -> per-lane results are 8 CONTIGUOUS elems of row b:
//                   vectorized pre loads (b128), packed bf16 h stores (b128), direct global x
//                   stores (b128). No barriers in the time loop (wave-private rows).
//   4) gemm_pre:    pre = Wih x x^T formulation, same contiguous-store trick.
//   5) proj:        out = x[-10:] @ Wp^T + bp  (tiny, VALU)
// Workspace layout (needs ~201.4 MB): [pre f32 R*128][x bf16 R*128][table f32 10*128]

typedef __attribute__((ext_vector_type(16))) __bf16 v16bf;
typedef __attribute__((ext_vector_type(8)))  __bf16 bf16x8;
typedef __attribute__((ext_vector_type(8)))  float  v8f;
typedef __attribute__((ext_vector_type(4)))  float  f32x4;

#define HDIM 128
#define TSTEPS 1024
#define BATCH 256

__device__ __forceinline__ v16bf make_frag(bf16x8 lo, bf16x8 hi) {
  return __builtin_shufflevector(lo, hi, 0,1,2,3,4,5,6,7,8,9,10,11,12,13,14,15);
}

// A fragment (16xK slice of 16-bit A, K-tile at k0), row-major source:
// lanes 0-15 (row m=lane): K {k0..k0+7} in v0-3, {k0+16..k0+23} in v4-7
// lanes 16-31 (row m=lane-16): K {k0+8..k0+15}, {k0+24..k0+31}
__device__ __forceinline__ v16bf load_fragA(const __bf16* base, int stride, int row,
                                            int k0, int lane) {
  const __bf16* p = base + (size_t)row * stride + k0 + ((lane >> 4) << 3);
  return make_frag(*(const bf16x8*)p, *(const bf16x8*)(p + 16));
}

// B fragment: B[k][n] sourced from row-major M[n][k] (i.e. B = M^T).
// lane n (0-15): K = k0..k0+7, k0+16..k0+23 of row n; lane n+16: K = k0+8..k0+15, k0+24..k0+31.
__device__ __forceinline__ v16bf load_fragB(const __bf16* base, int stride, int row,
                                            int k0, int lane) {
  const __bf16* p = base + (size_t)row * stride + k0 + ((lane >> 4) << 3);
  return make_frag(*(const bf16x8*)p, *(const bf16x8*)(p + 16));
}

// Branch-free tanh: tanh(x) = 1 - 2/(1 + exp2(x*2*log2(e))). Saturates correctly at +-inf.
__device__ __forceinline__ float fast_tanh(float x) {
  float e = __builtin_amdgcn_exp2f(x * 2.885390081777927f);
  float r = __builtin_amdgcn_rcpf(1.0f + e);
  return __builtin_fmaf(-2.0f, r, 1.0f);
}

// ---------------------------------------------------------------- fused embed+Wih0 LUT
__global__ void fuse_table(const float* __restrict__ embed, const float* __restrict__ Wih0,
                           const float* __restrict__ bih, const float* __restrict__ bhh,
                           float* __restrict__ table) {
  int i = threadIdx.x;  // 128
  float b = bih[i] + bhh[i];
  for (int v = 0; v < 10; ++v) {
    float acc = b;
    #pragma unroll
    for (int e = 0; e < 10; ++e) acc += embed[v * 10 + e] * Wih0[i * 10 + e];
    table[v * HDIM + i] = acc;
  }
}

// ---------------------------------------------------------------- layer-0 pre = gather
__global__ void gather_pre0(const int* __restrict__ tok, const float* __restrict__ table,
                            float* __restrict__ pre, int total) {
  int idx = blockIdx.x * 256 + threadIdx.x;
  if (idx < total) {
    int r = idx >> 7, i = idx & (HDIM - 1);
    pre[idx] = table[tok[r] * HDIM + i];
  }
}

// ---------------------------------------------------------------- recurrent scan (WMMA)
// Block: 256 threads = 8 waves; wave w owns chain rows [blockIdx*128 + 16w, +16) (the N axis).
// Per step: B frags = own h rows (LDS), A frags = Whh (loop-invariant, register-hoisted),
// D element (i, b) -> lane holds 8 contiguous i of its own row b.
__global__ void __launch_bounds__(256)
rnn_scan(const float* __restrict__ pre, const float* __restrict__ Whh,
         __bf16* __restrict__ xout, int N, int Tc) {
  __shared__ __bf16 sW[HDIM * HDIM];   // Whh row-major [i][j]  (A source)
  __shared__ __bf16 sH[HDIM * HDIM];   // h   row-major [b][j]  (B source)
  const int tid = threadIdx.x, wave = tid >> 5, lane = tid & 31;
  for (int i = tid; i < HDIM * HDIM; i += 256) {
    sW[i] = (__bf16)Whh[i];
    sH[i] = (__bf16)0.0f;   // h0 = 0
  }
  __syncthreads();
  const int m0b = wave * 16;                 // wave's chain-row tile (N axis)
  const int rowbase = blockIdx.x * 128;
  const int lr = lane & 15, hi = lane >> 4;
  const int coff = 8 * hi;                   // per-lane offset along i (M axis)
  const int brow = m0b + lr;                 // this lane's local chain row

  for (int t = 0; t < Tc; ++t) {
    // B fragments: h row b (own row), full K=128
    v16bf b0 = load_fragB(sH, HDIM, brow,  0, lane);
    v16bf b1 = load_fragB(sH, HDIM, brow, 32, lane);
    v16bf b2 = load_fragB(sH, HDIM, brow, 64, lane);
    v16bf b3 = load_fragB(sH, HDIM, brow, 96, lane);

    const size_t prow = (size_t)t * N + rowbase + brow;   // this lane's global row
    #pragma unroll
    for (int it = 0; it < 8; ++it) {
      // vectorized pre loads: 8 contiguous floats of row b (issued before the WMMA chain)
      const float* pb = pre + prow * HDIM + it * 16 + coff;
      f32x4 p0 = *(const f32x4*)pb;
      f32x4 p1 = *(const f32x4*)(pb + 4);

      v8f c = {};
      c = __builtin_amdgcn_wmma_f32_16x16x32_bf16(false,
            load_fragA(sW, HDIM, it * 16 + lr,  0, lane), false, b0, (short)0, c, false, false);
      c = __builtin_amdgcn_wmma_f32_16x16x32_bf16(false,
            load_fragA(sW, HDIM, it * 16 + lr, 32, lane), false, b1, (short)0, c, false, false);
      c = __builtin_amdgcn_wmma_f32_16x16x32_bf16(false,
            load_fragA(sW, HDIM, it * 16 + lr, 64, lane), false, b2, (short)0, c, false, false);
      c = __builtin_amdgcn_wmma_f32_16x16x32_bf16(false,
            load_fragA(sW, HDIM, it * 16 + lr, 96, lane), false, b3, (short)0, c, false, false);

      // tanh, pack 8 contiguous bf16, one LDS store + one direct global store
      bf16x8 hv;
      #pragma unroll
      for (int v = 0; v < 4; ++v) hv[v]     = (__bf16)fast_tanh(c[v]     + p0[v]);
      #pragma unroll
      for (int v = 0; v < 4; ++v) hv[4 + v] = (__bf16)fast_tanh(c[4 + v] + p1[v]);
      *(bf16x8*)&sH[(size_t)brow * HDIM + it * 16 + coff] = hv;
      *(bf16x8*)&xout[prow * HDIM + it * 16 + coff]       = hv;
    }
  }
}

// ---------------------------------------------------------------- pre = x @ Wih^T + b
// Same swapped formulation: A = Wih (LDS), B = x rows (global b128 loads),
// D element (i, r) -> contiguous f32x4 stores of pre row r.
__global__ void __launch_bounds__(256)
gemm_pre(const __bf16* __restrict__ x, const float* __restrict__ Wih,
         const float* __restrict__ bih, const float* __restrict__ bhh,
         float* __restrict__ pre) {
  __shared__ __bf16 sW[HDIM * HDIM];
  __shared__ float sb[HDIM];
  const int tid = threadIdx.x, wave = tid >> 5, lane = tid & 31;
  for (int i = tid; i < HDIM * HDIM; i += 256) sW[i] = (__bf16)Wih[i];
  if (tid < HDIM) sb[tid] = bih[tid] + bhh[tid];
  __syncthreads();
  const int lr = lane & 15, hi = lane >> 4;
  const int coff = 8 * hi;
  const size_t prow = (size_t)blockIdx.x * 128 + wave * 16 + lr;  // this lane's row of x/pre

  // B fragments: x row r straight from global (contiguous b128 chunks)
  v16bf b0 = load_fragB(x, HDIM, (int)prow,  0, lane);
  v16bf b1 = load_fragB(x, HDIM, (int)prow, 32, lane);
  v16bf b2 = load_fragB(x, HDIM, (int)prow, 64, lane);
  v16bf b3 = load_fragB(x, HDIM, (int)prow, 96, lane);

  #pragma unroll
  for (int it = 0; it < 8; ++it) {
    f32x4 s0 = *(const f32x4*)&sb[it * 16 + coff];
    f32x4 s1 = *(const f32x4*)&sb[it * 16 + coff + 4];
    v8f c;
    #pragma unroll
    for (int v = 0; v < 4; ++v) { c[v] = s0[v]; c[4 + v] = s1[v]; }
    c = __builtin_amdgcn_wmma_f32_16x16x32_bf16(false,
          load_fragA(sW, HDIM, it * 16 + lr,  0, lane), false, b0, (short)0, c, false, false);
    c = __builtin_amdgcn_wmma_f32_16x16x32_bf16(false,
          load_fragA(sW, HDIM, it * 16 + lr, 32, lane), false, b1, (short)0, c, false, false);
    c = __builtin_amdgcn_wmma_f32_16x16x32_bf16(false,
          load_fragA(sW, HDIM, it * 16 + lr, 64, lane), false, b2, (short)0, c, false, false);
    c = __builtin_amdgcn_wmma_f32_16x16x32_bf16(false,
          load_fragA(sW, HDIM, it * 16 + lr, 96, lane), false, b3, (short)0, c, false, false);
    float* pp = pre + prow * HDIM + it * 16 + coff;
    f32x4 o0, o1;
    #pragma unroll
    for (int v = 0; v < 4; ++v) { o0[v] = c[v]; o1[v] = c[4 + v]; }
    *(f32x4*)pp       = o0;
    *(f32x4*)(pp + 4) = o1;
  }
}

// ---------------------------------------------------------------- final projection
__global__ void proj(const __bf16* __restrict__ x, const float* __restrict__ Wp,
                     const float* __restrict__ bp, float* __restrict__ out) {
  int idx = blockIdx.x * 256 + threadIdx.x;     // 10*256*8 = 20480
  int o = idx & 7, rb = idx >> 3;               // rb = s*256 + b, s in 0..9
  const __bf16* xr = x + ((size_t)(TSTEPS - 10) * BATCH + rb) * HDIM;
  float acc = bp[o];
  #pragma unroll 8
  for (int k = 0; k < HDIM; ++k) acc += (float)xr[k] * Wp[o * HDIM + k];
  out[idx] = acc;
}

// ---------------------------------------------------------------- host launcher
extern "C" void kernel_launch(void* const* d_in, const int* in_sizes, int n_in,
                              void* d_out, int out_size, void* d_ws, size_t ws_size,
                              hipStream_t stream) {
  const int*   tok   = (const int*)d_in[0];
  const float* embed = (const float*)d_in[1];
  const float* Wih0  = (const float*)d_in[2];
  const float* Wih   = (const float*)d_in[3];
  const float* Whh   = (const float*)d_in[4];
  const float* bih   = (const float*)d_in[5];
  const float* bhh   = (const float*)d_in[6];
  const float* Wp    = (const float*)d_in[7];
  const float* bp    = (const float*)d_in[8];
  float* out = (float*)d_out;

  const size_t R = (size_t)TSTEPS * BATCH;            // 262144 rows
  char* ws = (char*)d_ws;
  float*  pre   = (float*)ws;                          // R*128 f32 = 128 MiB
  __bf16* x     = (__bf16*)(ws + R * HDIM * sizeof(float));
  float*  table = (float*)(ws + R * HDIM * sizeof(float) + R * HDIM * sizeof(__bf16));

  const int dil[5] = {1, 2, 4, 8, 16};

  fuse_table<<<1, 128, 0, stream>>>(embed, Wih0, bih, bhh, table);
  gather_pre0<<<(int)(R * HDIM / 256), 256, 0, stream>>>(tok, table, pre, (int)(R * HDIM));
  rnn_scan<<<dil[0] * BATCH / 128, 256, 0, stream>>>(pre, Whh, x,
                                                     dil[0] * BATCH, TSTEPS / dil[0]);
  for (int l = 1; l < 5; ++l) {
    gemm_pre<<<(int)(R / 128), 256, 0, stream>>>(
        x, Wih + (size_t)(l - 1) * HDIM * HDIM, bih + l * HDIM, bhh + l * HDIM, pre);
    rnn_scan<<<dil[l] * BATCH / 128, 256, 0, stream>>>(
        pre, Whh + (size_t)l * HDIM * HDIM, x, dil[l] * BATCH, TSTEPS / dil[l]);
  }
  proj<<<(10 * BATCH * 8) / 256, 256, 0, stream>>>(x, Wp, bp, out);
}
